// LLamaA16MlpBlock_60181081751677
// MI455X (gfx1250) — matrix operand order repaired and checked by reading, not compiled, not run on hardware
//
#include <hip/hip_runtime.h>

// ---------------------------------------------------------------------------
// LLaMA A16W4 MLP block for MI455X (gfx1250, wave32, WMMA 16x16x32 f16)
//   g = x @ dequant(qw0); u = x @ dequant(qw1); h = silu(g)*u (f16 -> ws)
//   out = h @ dequant(qw2)                                    (f32 -> d_out)
// ---------------------------------------------------------------------------

typedef __attribute__((ext_vector_type(16))) _Float16 v16h;
typedef __attribute__((ext_vector_type(8)))  _Float16 v8h;
typedef __attribute__((ext_vector_type(4)))  _Float16 v4h;
typedef __attribute__((ext_vector_type(8)))  float    v8f;
typedef __attribute__((ext_vector_type(4)))  int      v4i;

#define HDIM 4096
#define IDIM 11008
#define MTOK 2048
#define BM 128
#define BN 128
#define BK 32
#define LDP (BK + 8)   // 40 halves = 80B row stride: 16B aligned, bank-friendly

#define GLOBAL_AS __attribute__((address_space(1)))
#define LDS_AS    __attribute__((address_space(3)))

#if defined(__gfx1250__) && \
    __has_builtin(__builtin_amdgcn_global_load_async_to_lds_b128) && \
    __has_builtin(__builtin_amdgcn_s_wait_asynccnt)
#define USE_ASYNC_LDS 1
#else
#define USE_ASYNC_LDS 0
#endif

// A fragment (16x32 f16): lane r=lane&15 is row M=r; half=lane>>4.
// v16h elements 0..7  = K[half*8    .. half*8+7]
// v16h elements 8..15 = K[16+half*8 .. 16+half*8+7]
__device__ __forceinline__ v16h load_a_frag(const _Float16* row, int half) {
    v8h lo = *(const v8h*)(row + half * 8);
    v8h hi = *(const v8h*)(row + 16 + half * 8);
    return __builtin_shufflevector(lo, hi, 0, 1, 2, 3, 4, 5, 6, 7,
                                           8, 9, 10, 11, 12, 13, 14, 15);
}

// B fragment (32x16 f16), LDS stored transposed as Bt[n][k]:
// lane r = column N; elements 0..7 = K[half*8..+7], 8..15 = K[16+half*8..+7].
__device__ __forceinline__ v16h load_b_frag(const _Float16* row, int half) {
    v8h lo = *(const v8h*)(row + half * 8);
    v8h hi = *(const v8h*)(row + 16 + half * 8);
    return __builtin_shufflevector(lo, hi, 0, 1, 2, 3, 4, 5, 6, 7,
                                           8, 9, 10, 11, 12, 13, 14, 15);
}

// ---------------------------------------------------------------------------
// Kernel 1: fused gate+up GEMM + SwiGLU.  grid = (M/BM, I/BN), 256 threads.
// blockIdx.x = M block (fast dim) so workgroups sharing a weight N-strip are
// co-scheduled and the qw strip stays L2-resident (read-once from HBM).
// ---------------------------------------------------------------------------
__global__ __launch_bounds__(256, 1)
void gateup_swiglu_kernel(const float* __restrict__ x,
                          const int* __restrict__ qw0, const int* __restrict__ zp0,
                          const float* __restrict__ s0,
                          const int* __restrict__ qw1, const int* __restrict__ zp1,
                          const float* __restrict__ s1,
                          _Float16* __restrict__ hbuf) {
    __shared__ __align__(16) _Float16 As[BM * LDP];
    __shared__ __align__(16) _Float16 Bg[BN * LDP];   // transposed: [n][k]
    __shared__ __align__(16) _Float16 Bu[BN * LDP];

    const int t    = threadIdx.x;
    const int wid  = t >> 5;
    const int lane = t & 31;
    const int half = lane >> 4;
    const int r    = lane & 15;

    const int m_blk = blockIdx.x * BM;
    const int n_blk = blockIdx.y * BN;
    const int m_off = (wid >> 1) * 32;   // 4 wave-rows of M
    const int n_off = (wid & 1) * 64;    // 2 wave-cols of N

    // B-staging mapping: this thread owns 4 k x 4 n of each weight tile.
    const int kb = (t >> 5) * 4;         // 0,4,...,28
    const int nq = (t & 31) * 4;         // 0,4,...,124

    v8f accg[2][4] = {};
    v8f accu[2][4] = {};

    for (int k0 = 0; k0 < HDIM; k0 += BK) {
        const int grp = k0 >> 7;   // group constant within a BK=32 tile

        // ---- stage A tile (128x32 fp32 -> f16), coalesced float4 ----
#pragma unroll
        for (int p = 0; p < 4; ++p) {
            int idx = t + p * 256;
            int row = idx >> 3;
            int ks  = (idx & 7) * 4;
            float4 v = *(const float4*)(x + (size_t)(m_blk + row) * HDIM + k0 + ks);
            v4h h;
            h[0] = (_Float16)v.x; h[1] = (_Float16)v.y;
            h[2] = (_Float16)v.z; h[3] = (_Float16)v.w;
            *(v4h*)&As[row * LDP + ks] = h;
        }

        // ---- stage + dequant both weight tiles (32k x 128n), transposed ----
        {
            const size_t nbase = (size_t)n_blk + nq;
            const size_t zoff  = (size_t)grp * IDIM + nbase;
            int4   z0 = *(const int4*)(zp0 + zoff);
            float4 c0 = *(const float4*)(s0 + zoff);
            int4   z1 = *(const int4*)(zp1 + zoff);
            float4 c1 = *(const float4*)(s1 + zoff);
            // precompute -zp*scale per column
            float g0x = -(float)z0.x * c0.x, g0y = -(float)z0.y * c0.y;
            float g0z = -(float)z0.z * c0.z, g0w = -(float)z0.w * c0.w;
            float g1x = -(float)z1.x * c1.x, g1y = -(float)z1.y * c1.y;
            float g1z = -(float)z1.z * c1.z, g1w = -(float)z1.w * c1.w;

            v4h ga, gb, gc, gd;   // gate cols n = nq+0..3, k = kb..kb+3
            v4h ua, ub, uc, ud;   // up   cols
#pragma unroll
            for (int j = 0; j < 4; ++j) {
                size_t goff = (size_t)(k0 + kb + j) * IDIM + nbase;
                int4 q = *(const int4*)(qw0 + goff);
                ga[j] = (_Float16)fmaf((float)q.x, c0.x, g0x);
                gb[j] = (_Float16)fmaf((float)q.y, c0.y, g0y);
                gc[j] = (_Float16)fmaf((float)q.z, c0.z, g0z);
                gd[j] = (_Float16)fmaf((float)q.w, c0.w, g0w);
                int4 p = *(const int4*)(qw1 + goff);
                ua[j] = (_Float16)fmaf((float)p.x, c1.x, g1x);
                ub[j] = (_Float16)fmaf((float)p.y, c1.y, g1y);
                uc[j] = (_Float16)fmaf((float)p.z, c1.z, g1z);
                ud[j] = (_Float16)fmaf((float)p.w, c1.w, g1w);
            }
            *(v4h*)&Bg[(nq + 0) * LDP + kb] = ga;
            *(v4h*)&Bg[(nq + 1) * LDP + kb] = gb;
            *(v4h*)&Bg[(nq + 2) * LDP + kb] = gc;
            *(v4h*)&Bg[(nq + 3) * LDP + kb] = gd;
            *(v4h*)&Bu[(nq + 0) * LDP + kb] = ua;
            *(v4h*)&Bu[(nq + 1) * LDP + kb] = ub;
            *(v4h*)&Bu[(nq + 2) * LDP + kb] = uc;
            *(v4h*)&Bu[(nq + 3) * LDP + kb] = ud;

            if (k0 + BK < HDIM) {   // warm L2 for the next k-step's weight rows
                size_t pf = (size_t)(k0 + BK + kb) * IDIM + nbase;
                __builtin_prefetch(qw0 + pf, 0, 0);
                __builtin_prefetch(qw1 + pf, 0, 0);
            }
        }
        __syncthreads();

        // ---- WMMA: 2 (M) x 4 (N) tiles per wave, both matrices ----
        v16h af[2];
        af[0] = load_a_frag(&As[(m_off + r) * LDP], half);
        af[1] = load_a_frag(&As[(m_off + 16 + r) * LDP], half);
#pragma unroll
        for (int nt = 0; nt < 4; ++nt) {
            v16h bg = load_b_frag(&Bg[(n_off + nt * 16 + r) * LDP], half);
            v16h bu = load_b_frag(&Bu[(n_off + nt * 16 + r) * LDP], half);
#pragma unroll
            for (int mt = 0; mt < 2; ++mt) {
                accg[mt][nt] = __builtin_amdgcn_wmma_f32_16x16x32_f16(
                    false, af[mt], false, bg, (short)0, accg[mt][nt], false, false);
                accu[mt][nt] = __builtin_amdgcn_wmma_f32_16x16x32_f16(
                    false, af[mt], false, bu, (short)0, accu[mt][nt], false, false);
            }
        }
        __syncthreads();
    }

    // ---- epilogue: h = silu(g) * u, stored f16 ----
    // C/D layout: n = lane&15 (+16..31 same), m = vgpr + 8*(lane>=16)
#pragma unroll
    for (int mt = 0; mt < 2; ++mt) {
#pragma unroll
        for (int nt = 0; nt < 4; ++nt) {
            int n = n_blk + n_off + nt * 16 + r;
#pragma unroll
            for (int v = 0; v < 8; ++v) {
                int m = m_blk + m_off + mt * 16 + v + half * 8;
                float g = accg[mt][nt][v];
                float u = accu[mt][nt][v];
                float h = (g / (1.0f + __expf(-g))) * u;
                hbuf[(size_t)m * IDIM + n] = (_Float16)h;
            }
        }
    }
}

// ---------------------------------------------------------------------------
// Kernel 2: down projection.  grid = (M/BM, HDIM/BN), 256 threads.
// A = h (f16, K=11008), B = dequant(qw2), output fp32.
// A staging is a pure f16 copy -> use CDNA5 async global->LDS loads
// (ASYNCcnt-tracked, no VGPR round trip) when the builtin is available.
// ---------------------------------------------------------------------------
__global__ __launch_bounds__(256, 1)
void down_proj_kernel(const _Float16* __restrict__ hbuf,
                      const int* __restrict__ qw2, const int* __restrict__ zp2,
                      const float* __restrict__ s2,
                      float* __restrict__ out) {
    __shared__ __align__(16) _Float16 As[BM * LDP];
    __shared__ __align__(16) _Float16 Bd[BN * LDP];   // transposed: [n][k]

    const int t    = threadIdx.x;
    const int wid  = t >> 5;
    const int lane = t & 31;
    const int half = lane >> 4;
    const int r    = lane & 15;

    const int m_blk = blockIdx.x * BM;
    const int n_blk = blockIdx.y * BN;
    const int m_off = (wid >> 1) * 32;
    const int n_off = (wid & 1) * 64;

    const int kb = (t >> 5) * 4;
    const int nq = (t & 31) * 4;

    v8f acc[2][4] = {};

    for (int k0 = 0; k0 < IDIM; k0 += BK) {
        const int grp = k0 >> 7;

        // ---- stage A tile (128x32 f16): async copy global -> LDS ----
#pragma unroll
        for (int p = 0; p < 2; ++p) {
            int idx = t + p * 256;
            int row = idx >> 2;
            int ks  = (idx & 3) * 8;
            const _Float16* g = hbuf + (size_t)(m_blk + row) * IDIM + k0 + ks;
            _Float16* l = &As[row * LDP + ks];
#if USE_ASYNC_LDS
            // param0: generic int4* to global src; param1: AS(3) int4* LDS dst;
            // then imm offset, imm cpol (arity 4 confirmed by probe).
            __builtin_amdgcn_global_load_async_to_lds_b128(
                (v4i*)g, (LDS_AS v4i*)l, 0, 0);
#else
            *(v8h*)l = *(const v8h*)g;
#endif
        }

        // ---- stage + dequant weight tile (4k x 4n per thread), transposed ----
        {
            const size_t nbase = (size_t)n_blk + nq;
            const size_t zoff  = (size_t)grp * HDIM + nbase;
            int4   z = *(const int4*)(zp2 + zoff);
            float4 c = *(const float4*)(s2 + zoff);
            float gx = -(float)z.x * c.x, gy = -(float)z.y * c.y;
            float gz = -(float)z.z * c.z, gw = -(float)z.w * c.w;

            v4h da, db, dc, dd;
#pragma unroll
            for (int j = 0; j < 4; ++j) {
                size_t goff = (size_t)(k0 + kb + j) * HDIM + nbase;
                int4 q = *(const int4*)(qw2 + goff);
                da[j] = (_Float16)fmaf((float)q.x, c.x, gx);
                db[j] = (_Float16)fmaf((float)q.y, c.y, gy);
                dc[j] = (_Float16)fmaf((float)q.z, c.z, gz);
                dd[j] = (_Float16)fmaf((float)q.w, c.w, gw);
            }
            *(v4h*)&Bd[(nq + 0) * LDP + kb] = da;
            *(v4h*)&Bd[(nq + 1) * LDP + kb] = db;
            *(v4h*)&Bd[(nq + 2) * LDP + kb] = dc;
            *(v4h*)&Bd[(nq + 3) * LDP + kb] = dd;

            if (k0 + BK < IDIM) {
                __builtin_prefetch(qw2 + (size_t)(k0 + BK + kb) * HDIM + nbase, 0, 0);
            }
        }
#if USE_ASYNC_LDS
        __builtin_amdgcn_s_wait_asynccnt(0);
#endif
        __syncthreads();

        v16h af[2];
        af[0] = load_a_frag(&As[(m_off + r) * LDP], half);
        af[1] = load_a_frag(&As[(m_off + 16 + r) * LDP], half);
#pragma unroll
        for (int nt = 0; nt < 4; ++nt) {
            v16h b = load_b_frag(&Bd[(n_off + nt * 16 + r) * LDP], half);
#pragma unroll
            for (int mt = 0; mt < 2; ++mt) {
                acc[mt][nt] = __builtin_amdgcn_wmma_f32_16x16x32_f16(
                    false, af[mt], false, b, (short)0, acc[mt][nt], false, false);
            }
        }
        __syncthreads();
    }

#pragma unroll
    for (int mt = 0; mt < 2; ++mt) {
#pragma unroll
        for (int nt = 0; nt < 4; ++nt) {
            int n = n_blk + n_off + nt * 16 + r;
#pragma unroll
            for (int v = 0; v < 8; ++v) {
                int m = m_blk + m_off + mt * 16 + v + half * 8;
                out[(size_t)m * HDIM + n] = acc[mt][nt][v];
            }
        }
    }
}

// ---------------------------------------------------------------------------
extern "C" void kernel_launch(void* const* d_in, const int* in_sizes, int n_in,
                              void* d_out, int out_size, void* d_ws, size_t ws_size,
                              hipStream_t stream) {
    const float* x   = (const float*)d_in[0];
    const int*   qw0 = (const int*)d_in[1];
    const int*   zp0 = (const int*)d_in[2];
    const float* s0  = (const float*)d_in[3];
    const int*   qw1 = (const int*)d_in[4];
    const int*   zp1 = (const int*)d_in[5];
    const float* s1  = (const float*)d_in[6];
    const int*   qw2 = (const int*)d_in[7];
    const int*   zp2 = (const int*)d_in[8];
    const float* s2  = (const float*)d_in[9];
    // d_in[10] = group_size (always 128, baked into the kernels)

    _Float16* hbuf = (_Float16*)d_ws;          // 2048 x 11008 f16 = 45 MB
    float*    out  = (float*)d_out;            // 1 x 2048 x 4096 fp32

    dim3 grid1(MTOK / BM, IDIM / BN);          // (16, 86), M fastest for L2 reuse
    gateup_swiglu_kernel<<<grid1, 256, 0, stream>>>(x, qw0, zp0, s0,
                                                    qw1, zp1, s1, hbuf);

    dim3 grid2(MTOK / BM, HDIM / BN);          // (16, 32)
    down_proj_kernel<<<grid2, 256, 0, stream>>>(hbuf, qw2, zp2, s2, out);
}
